// UpsamplingNetwork_82033875354108
// MI455X (gfx1250) — matrix operand order repaired and checked by reading, not compiled, not run on hardware
//
#include <hip/hip_runtime.h>

// ---------------------------------------------------------------------------
// Types for CDNA5 WMMA (gfx1250, wave32)
// ---------------------------------------------------------------------------
typedef __attribute__((ext_vector_type(16))) __bf16 v16bf;
typedef __attribute__((ext_vector_type(8)))  float  v8f;

struct Q2 { uint4 a, b; };   // 32 bytes == v16bf

__device__ __forceinline__ unsigned short f2bfbits(float f) {
  unsigned u = __builtin_bit_cast(unsigned, f);
  return (unsigned short)((u + 0x7FFFu + ((u >> 16) & 1u)) >> 16);
}
// Exact small-range division via fp32 reciprocal + one fixup (no v_rcp chains).
__device__ __forceinline__ int fdiv(int n, int d, float inv, int& rem) {
  int q = (int)((float)n * inv);
  int r = n - q * d;
  if (r < 0)       { q -= 1; r += d; }
  else if (r >= d) { q += 1; r -= d; }
  rem = r;
  return q;
}
__device__ __forceinline__ v16bf ldfrag(const uint4* p) {   // 2x b128
  Q2 t; t.a = p[0]; t.b = p[1];
  return __builtin_bit_cast(v16bf, t);
}

// ---------------------------------------------------------------------------
// fp32 -> bf16 conversion (external inputs)
// ---------------------------------------------------------------------------
__global__ void cvt_bf16_kernel(const float* __restrict__ src,
                                unsigned short* __restrict__ dst, int n) {
  int i = blockIdx.x * blockDim.x + threadIdx.x;
  if (i < n) dst[i] = f2bfbits(src[i]);
}

// ---------------------------------------------------------------------------
// Weight pre-pack: fp32 OIHW -> bf16 in per-lane WMMA A-fragment order, k = 16*(lane/16) + j (same as B).
// dst: [mtilesPad][nChunks][32 lanes][16 halfs]; grid = (2, nChunks, mtilesPad)
// ---------------------------------------------------------------------------
__global__ void pack_wmma_kernel(const float* __restrict__ src,
                                 unsigned short* __restrict__ dst,
                                 int Cout, int Ktot, int nChunks) {
  int il   = blockIdx.x * 256 + threadIdx.x;       // 0..511
  if (il >= 512) return;
  int kc   = blockIdx.y;
  int mt   = blockIdx.z;
  int j    = il & 15;
  int lane = il >> 4;
  int m    = mt * 16 + (lane & 15);
  int klo  = (lane >> 4) * 8;
  int kk   = kc * 32 + 2 * klo + j;
  float v  = (m < Cout && kk < Ktot) ? src[m * Ktot + kk] : 0.f;
  dst[((size_t)mt * nChunks + kc) * 512 + il] = f2bfbits(v);
}

// ---------------------------------------------------------------------------
// Implicit-GEMM convolution via V_WMMA_F32_16X16X32_BF16.
//   Input activations: bf16. Block = 128 threads (4 waves), tile 64Cout x 64pix.
//   Each wave: one 16-row Cout tile x 4 accumulators (4 wmma / K-chunk).
//   B chunk (32K x 64N bf16, 4KB) staged in LDS, reused by all 4 waves.
//   A fragments: pre-packed -> two coalesced b128 loads per chunk per lane.
//   Fused: input relu (template, 1-op on bf16 sign), bias, output relu,
//   fp32 residual add; dual store fp32 (nullable) + bf16 mirror (nullable).
// ---------------------------------------------------------------------------
template <int KSIZE, bool RELU_IN>
__global__ __launch_bounds__(128)
void conv_wmma_kernel(const unsigned short* __restrict__ in,   // bf16 bits
                      const unsigned short* __restrict__ wt,   // packed bf16
                      const float* __restrict__ bias,
                      const float* __restrict__ resid,         // nullable fp32
                      float* __restrict__ out_f,               // nullable
                      unsigned short* __restrict__ out_bf,     // nullable
                      int Cin, int Cout, int H, int W, int NP,
                      float invHW, float invW, int relu_out) {
  constexpr int TAPS = KSIZE * KSIZE;
  constexpr int PAD  = (KSIZE - 1) / 2;
  const int HW   = H * W;
  const int Ktot = Cin * TAPS;
  const int nChunks = (Ktot + 31) >> 5;

  const int nt    = blockIdx.x;                 // 64-pixel N tile
  const int tid   = threadIdx.x;
  const int lane  = tid & 31;
  const int wave  = tid >> 5;
  const int mtile = blockIdx.y * 4 + wave;      // 16-row Cout tile (zero-padded)

  __shared__ unsigned short Bsh[64 * 32];       // [n][k] bf16 bits

  // Cooperative B fill: thread owns pixel n = tid&63, k-range kHalf..kHalf+15
  const int nB    = tid & 63;
  const int kHalf = (tid >> 6) * 16;
  const int pixB  = nt * 64 + nB;
  int bB = 0, yB = 0, xB = 0;
  const bool pixOK = pixB < NP;
  if (pixOK) {
    int r;
    bB = fdiv(pixB, HW, invHW, r);
    yB = fdiv(r, W, invW, xB);
  }
  const unsigned short* inb = in + (size_t)bB * Cin * HW;

  const unsigned short* wtile = wt + (size_t)mtile * nChunks * 512;

  v8f acc0 = {0.f,0.f,0.f,0.f,0.f,0.f,0.f,0.f};
  v8f acc1 = acc0, acc2 = acc0, acc3 = acc0;

  const int nl  = lane & 15;
  const int kb0 = (lane >> 4) * 16;

  auto loadB = [&](int kidx) -> unsigned {
    unsigned v = 0;
    if (pixOK && kidx < Ktot) {
      const int cin = kidx / TAPS;              // compile-time divisor
      const int t   = kidx - cin * TAPS;
      const int ky  = t / KSIZE;
      const int kx  = t - ky * KSIZE;
      const int yy  = yB + ky - PAD;
      const int xx  = xB + kx - PAD;
      if (yy >= 0 && yy < H && xx >= 0 && xx < W) {
        v = inb[cin * HW + yy * W + xx];
        if (RELU_IN) v = (v & 0x8000u) ? 0u : v;   // bf16 relu on sign bit
      }
    }
    return v;
  };

  for (int kc = 0; kc < nChunks; ++kc) {
    const int k0 = kc << 5;
    __syncthreads();
    // ---- stage B chunk into LDS (bf16 copy), two b128 LDS stores ----
    {
      unsigned p0 = loadB(k0+kHalf+ 0) | (loadB(k0+kHalf+ 1) << 16);
      unsigned p1 = loadB(k0+kHalf+ 2) | (loadB(k0+kHalf+ 3) << 16);
      unsigned p2 = loadB(k0+kHalf+ 4) | (loadB(k0+kHalf+ 5) << 16);
      unsigned p3 = loadB(k0+kHalf+ 6) | (loadB(k0+kHalf+ 7) << 16);
      unsigned p4 = loadB(k0+kHalf+ 8) | (loadB(k0+kHalf+ 9) << 16);
      unsigned p5 = loadB(k0+kHalf+10) | (loadB(k0+kHalf+11) << 16);
      unsigned p6 = loadB(k0+kHalf+12) | (loadB(k0+kHalf+13) << 16);
      unsigned p7 = loadB(k0+kHalf+14) | (loadB(k0+kHalf+15) << 16);
      uint4 q0; q0.x = p0; q0.y = p1; q0.z = p2; q0.w = p3;
      uint4 q1; q1.x = p4; q1.y = p5; q1.z = p6; q1.w = p7;
      uint4* bs = (uint4*)(Bsh + nB * 32 + kHalf);
      bs[0] = q0;
      bs[1] = q1;
    }
    __syncthreads();

    // ---- A fragment: two coalesced b128 loads from packed weights ----
    v16bf Av = ldfrag((const uint4*)(wtile + ((size_t)kc * 32 + lane) * 16));
    __builtin_prefetch(wtile + ((size_t)(kc + 1) * 32 + lane) * 16, 0, 1);

    // ---- 4 N-subtiles: B frag (2x ds_load_b128) interleaved with wmma ----
    const uint4* bbase = (const uint4*)(Bsh + nl * 32 + kb0);
    v16bf B0 = ldfrag(bbase);
    acc0 = __builtin_amdgcn_wmma_f32_16x16x32_bf16(false, Av, false, B0, (short)0, acc0, false, false);
    v16bf B1 = ldfrag(bbase + (16 * 32) / 8);
    acc1 = __builtin_amdgcn_wmma_f32_16x16x32_bf16(false, Av, false, B1, (short)0, acc1, false, false);
    v16bf B2 = ldfrag(bbase + (32 * 32) / 8);
    acc2 = __builtin_amdgcn_wmma_f32_16x16x32_bf16(false, Av, false, B2, (short)0, acc2, false, false);
    v16bf B3 = ldfrag(bbase + (48 * 32) / 8);
    acc3 = __builtin_amdgcn_wmma_f32_16x16x32_bf16(false, Av, false, B3, (short)0, acc3, false, false);
  }

  // ---- store: VGPR i -> M = i + 8*(lane>=16), N = lane&15 ----
  const int mhi = (lane >> 4) * 8;
  auto storeTile = [&](v8f a, int s) {
    const int pix = nt * 64 + s * 16 + nl;
    if (pix < NP) {
      int r2;
      const int b = fdiv(pix, HW, invHW, r2);
#pragma unroll
      for (int i = 0; i < 8; ++i) {
        const int cout = mtile * 16 + mhi + i;
        if (cout < Cout) {
          float v = a[i] + bias[cout];
          if (relu_out) v = fmaxf(v, 0.f);
          const int o = (b * Cout + cout) * HW + r2;
          if (resid) v += resid[o];
          if (out_f)  out_f[o]  = v;
          if (out_bf) out_bf[o] = f2bfbits(v);
        }
      }
    }
  };
  storeTile(acc0, 0);
  storeTile(acc1, 1);
  storeTile(acc2, 2);
  storeTile(acc3, 3);
}

// ---------------------------------------------------------------------------
// Elementwise / per-pixel helper kernels (div-free index decode via grid dims)
// ---------------------------------------------------------------------------
__global__ void mul_kernel(const float* __restrict__ a, const float* __restrict__ b,
                           float* __restrict__ o, int n) {
  int i = blockIdx.x * blockDim.x + threadIdx.x;
  if (i < n) o[i] = a[i] * b[i];
}

// data viewed as [B, G, S, HW]; x = x - mean_over_G + 1/G (in place)
// grid = (HW/256, S, B)
__global__ void chan_norm_kernel(float* __restrict__ data, int G, int S, int HW) {
  int pix = blockIdx.x * blockDim.x + threadIdx.x;
  if (pix >= HW) return;
  int s = blockIdx.y, b = blockIdx.z;
  float sum = 0.f;
  for (int g = 0; g < G; ++g) sum += data[((b * G + g) * S + s) * HW + pix];
  float adj = (1.f - sum) / (float)G;
  for (int g = 0; g < G; ++g) data[((b * G + g) * S + s) * HW + pix] += adj;
}

// net[b,c,y,x] += sum_{7x7} fm * lf ; refresh bf16 mirror. grid = (HW/256, C, B)
__global__ void localfilter7_kernel(const float* __restrict__ fm, const float* __restrict__ lf,
                                    float* __restrict__ net, unsigned short* __restrict__ net_bf,
                                    int C, int H, int W, float invW) {
  int HW = H * W;
  int pix = blockIdx.x * blockDim.x + threadIdx.x;
  if (pix >= HW) return;
  int c = blockIdx.y, b = blockIdx.z;
  int x;
  int y = fdiv(pix, W, invW, x);
  float s = 0.f;
#pragma clang loop unroll(disable)
  for (int p = 0; p < 49; ++p) {            // keep VGPRs low: no unroll
    int py = p / 7, px = p - (p / 7) * 7;
    int yy = y + py - 3, xx = x + px - 3;
    if (yy >= 0 && yy < H && xx >= 0 && xx < W)
      s += fm[(b * C + c) * HW + yy * W + xx] * lf[(b * 49 + p) * HW + pix];
  }
  int idx = (b * C + c) * HW + pix;
  float nv = net[idx] + s;
  net[idx] = nv;
  net_bf[idx] = f2bfbits(nv);
}

// pixel shuffle r=2 on bf16: in [B, C2*4, H, W] -> out [B, C2, 2H, 2W]
// grid = (HoWo/256, C2, B)
__global__ void pixel_shuffle2_bf_kernel(const unsigned short* __restrict__ in,
                                         unsigned short* __restrict__ out,
                                         int C2, int H, int W, float invWo) {
  int Ho = H * 2, Wo = W * 2;
  int pix = blockIdx.x * blockDim.x + threadIdx.x;
  if (pix >= Ho * Wo) return;
  int c = blockIdx.y, b = blockIdx.z;
  int x2;
  int y2 = fdiv(pix, Wo, invWo, x2);
  int y = y2 >> 1, ry = y2 & 1;
  int x = x2 >> 1, rx = x2 & 1;
  out[((size_t)(b * C2 + c) * Ho + y2) * Wo + x2] =
      in[((size_t)(b * C2 * 4) + c * 4 + ry * 2 + rx) * (H * W) + y * W + x];
}

// Dynamic 5x5 local upsample (scale 4) + residual add -> d_out; grid = (HW/256, 3, B)
// kern viewed as [B, 25, 16, H, W]
__global__ void upsample_local_kernel(const float* __restrict__ lr,
                                      const float* __restrict__ kern,
                                      const float* __restrict__ rbuf,
                                      float* __restrict__ out,
                                      int H, int W, float invW) {
  int HW = H * W;
  int pix = blockIdx.x * blockDim.x + threadIdx.x;
  if (pix >= HW) return;
  int c = blockIdx.y, b = blockIdx.z;
  int x;
  int y = fdiv(pix, W, invW, x);
  float acc[16];
#pragma unroll
  for (int s = 0; s < 16; ++s) acc[s] = 0.f;
#pragma clang loop unroll(disable)
  for (int p = 0; p < 25; ++p) {            // forced no-unroll: avoid VGPR spill
    int py = p / 5, px = p - (p / 5) * 5;
    int yy = y + py - 2, xx = x + px - 2;
    float v = 0.f;
    if (yy >= 0 && yy < H && xx >= 0 && xx < W) v = lr[(b * 3 + c) * HW + yy * W + xx];
    const float* kp = kern + (size_t)((b * 25 + p) * 16) * HW + pix;
#pragma unroll
    for (int s = 0; s < 16; ++s) acc[s] += v * kp[(size_t)s * HW];
  }
  int Wo = W * 4;
  int base = (b * 3 + c) * (H * 4) * Wo;
#pragma unroll
  for (int s = 0; s < 16; ++s) {
    int sy = s >> 2, sx = s & 3;
    int o = base + (y * 4 + sy) * Wo + (x * 4 + sx);
    out[o] = acc[s] + rbuf[o];
  }
}

// ---------------------------------------------------------------------------
// Host orchestration
// ---------------------------------------------------------------------------
extern "C" void kernel_launch(void* const* d_in, const int* in_sizes, int n_in,
                              void* d_out, int out_size, void* d_ws, size_t ws_size,
                              hipStream_t stream) {
  (void)in_sizes; (void)n_in; (void)out_size; (void)ws_size;

  const int B = 2, H = 96, W = 96, HW = H * W;

  const float* lr = (const float*)d_in[0];
  const float* dk = (const float*)d_in[1];

  // ---- params (JAX tree order: sorted dict keys) ----
  int pi = 2;
  auto nf = [&]() { return (const float*)d_in[pi++]; };
  const float* init_w = nf(); const float* init_b = nf();
  const float *kbw[2], *kbb[2];
  for (int i = 0; i < 2; ++i) { kbw[i] = nf(); kbb[i] = nf(); }
  const float *kfew[3], *kfeb[3];
  for (int i = 0; i < 3; ++i) { kfew[i] = nf(); kfeb[i] = nf(); }
  struct Koala { const float *fw[2], *fb[2], *lw[2], *lb[2], *mw[2], *mb[2]; } ko[5];
  for (int k = 0; k < 5; ++k) {
    for (int i = 0; i < 2; ++i) { ko[k].fw[i] = nf(); ko[k].fb[i] = nf(); }
    for (int i = 0; i < 2; ++i) { ko[k].lw[i] = nf(); ko[k].lb[i] = nf(); }
    for (int i = 0; i < 2; ++i) { ko[k].mw[i] = nf(); ko[k].mb[i] = nf(); }
  }
  const float *rw[7][2], *rb[7][2];
  for (int k = 0; k < 7; ++k)
    for (int i = 0; i < 2; ++i) { rw[k][i] = nf(); rb[k][i] = nf(); }
  const float *sw[3], *sb[3];
  for (int i = 0; i < 3; ++i) { sw[i] = nf(); sb[i] = nf(); }

  // ---- workspace bump allocator ----
  size_t off = 0;
  auto alloc = [&](size_t bytes) -> void* {
    void* p = (char*)d_ws + off;
    off = (off + bytes + 255) & ~(size_t)255;
    return p;
  };

  unsigned short* wb_cur = (unsigned short*)alloc((size_t)2600000 * 2);
  auto pack = [&](const float* w, int Cout, int Cin, int ks) -> const unsigned short* {
    int Ktot = Cin * ks * ks;
    int nChunks = (Ktot + 31) >> 5;
    int mtilesPad = (((Cout + 15) >> 4) + 3) & ~3;
    unsigned short* dst = wb_cur;
    wb_cur += (size_t)mtilesPad * nChunks * 512;
    pack_wmma_kernel<<<dim3(2, nChunks, mtilesPad), 256, 0, stream>>>(w, dst, Cout, Ktot, nChunks);
    return dst;
  };

  typedef unsigned short u16;
  const size_t f64ch  = (size_t)B * 64 * HW * 4;   // fp32 64-ch feature map
  const size_t b64ch  = (size_t)B * 64 * HW * 2;   // bf16 64-ch feature map
  float* net   = (float*)alloc(f64ch);
  float* t1    = (float*)alloc(f64ch);
  float* t2    = (float*)alloc(f64ch);
  float* lfb   = (float*)alloc((size_t)B * 49 * HW * 4);
  float* big0  = (float*)alloc((size_t)B * 400 * HW * 4);
  float* rbuf  = (float*)alloc((size_t)B * 3 * 384 * 384 * 4);
  u16* net_bf  = (u16*)alloc(b64ch);
  u16* kf_bf   = (u16*)alloc(b64ch);
  u16* t0bf    = (u16*)alloc(b64ch);
  u16* t1bf    = (u16*)alloc(b64ch);
  u16* dk_bf   = (u16*)alloc((size_t)B * 400 * HW * 2);
  u16* lr_bf   = (u16*)alloc((size_t)B * 3 * HW * 2);
  u16* big1bf  = (u16*)alloc((size_t)B * 64 * 192 * 192 * 2);
  u16* big2bf  = (u16*)alloc((size_t)B * 64 * 192 * 192 * 2);

  // ---- pack all conv weights ----
  const u16* Winit = pack(init_w, 64, 3, 3);
  const u16* Wkb[2]  = { pack(kbw[0], 128, 64, 3), pack(kbw[1], 400, 128, 3) };
  const u16* Wkfe[3] = { pack(kfew[0], 64, 400, 3), pack(kfew[1], 64, 64, 3),
                         pack(kfew[2], 64, 64, 3) };
  const u16 *Wfp[5][2], *Wlf[5][2], *Wmp[5][2];
  for (int k = 0; k < 5; ++k) {
    for (int i = 0; i < 2; ++i) Wfp[k][i] = pack(ko[k].fw[i], 64, 64, 3);
    Wlf[k][0] = pack(ko[k].lw[0], 64, 64, 1);
    Wlf[k][1] = pack(ko[k].lw[1], 49, 64, 1);
    for (int i = 0; i < 2; ++i) Wmp[k][i] = pack(ko[k].mw[i], 64, 64, 3);
  }
  const u16* Wres[7][2];
  for (int k = 0; k < 7; ++k)
    for (int i = 0; i < 2; ++i) Wres[k][i] = pack(rw[k][i], 64, 64, 3);
  const u16* Wsd[3] = { pack(sw[0], 128, 64, 3), pack(sw[1], 64, 32, 3),
                        pack(sw[2], 3, 16, 3) };

  // ---- conv launcher ----
  auto conv = [&](const u16* in, const u16* w, const float* bias, const float* resid,
                  float* out_f, u16* out_bf,
                  int Cin, int Cout, int Hc, int Wc, int ks, int rin, int rout) {
    int NP = B * Hc * Wc;
    int mtilesPad = (((Cout + 15) >> 4) + 3) & ~3;
    dim3 grid((NP + 63) / 64, mtilesPad / 4);
    float invHW = 1.0f / (float)(Hc * Wc), invW = 1.0f / (float)Wc;
    if (ks == 1)
      conv_wmma_kernel<1, false><<<grid, 128, 0, stream>>>(in, w, bias, resid, out_f, out_bf,
          Cin, Cout, Hc, Wc, NP, invHW, invW, rout);
    else if (rin)
      conv_wmma_kernel<3, true><<<grid, 128, 0, stream>>>(in, w, bias, resid, out_f, out_bf,
          Cin, Cout, Hc, Wc, NP, invHW, invW, rout);
    else
      conv_wmma_kernel<3, false><<<grid, 128, 0, stream>>>(in, w, bias, resid, out_f, out_bf,
          Cin, Cout, Hc, Wc, NP, invHW, invW, rout);
  };
  auto gp = [&](int hw) { return (hw + 255) / 256; };

  // ---- external inputs to bf16 ----
  cvt_bf16_kernel<<<dim3((B*400*HW + 255)/256), 256, 0, stream>>>(dk, dk_bf, B*400*HW);
  cvt_bf16_kernel<<<dim3((B*3*HW + 255)/256), 256, 0, stream>>>(lr, lr_bf, B*3*HW);

  // ---- kernel feature extractor: (conv3x3 -> relu) x3 ----
  conv(dk_bf, Wkfe[0], kfeb[0], nullptr, nullptr, t0bf, 400, 64, H, W, 3, 0, 1);
  conv(t0bf,  Wkfe[1], kfeb[1], nullptr, nullptr, t1bf, 64, 64, H, W, 3, 0, 1);
  conv(t1bf,  Wkfe[2], kfeb[2], nullptr, nullptr, kf_bf, 64, 64, H, W, 3, 0, 1);

  // ---- init conv: fp32 spine + bf16 mirror ----
  conv(lr_bf, Winit, init_b, nullptr, net, net_bf, 3, 64, H, W, 3, 0, 0);

  // ---- 5 KOALA blocks ----
  for (int k = 0; k < 5; ++k) {
    conv(net_bf, Wfp[k][0], ko[k].fb[0], nullptr, nullptr, t0bf, 64, 64, H, W, 3, 1, 1);
    conv(t0bf,   Wfp[k][1], ko[k].fb[1], nullptr, t1, nullptr, 64, 64, H, W, 3, 0, 0);  // fp
    conv(kf_bf,  Wmp[k][0], ko[k].mb[0], nullptr, nullptr, t0bf, 64, 64, H, W, 3, 0, 1);
    conv(t0bf,   Wmp[k][1], ko[k].mb[1], nullptr, t2, nullptr, 64, 64, H, W, 3, 0, 0);  // mp
    conv(kf_bf,  Wlf[k][0], ko[k].lb[0], nullptr, nullptr, t0bf, 64, 64, H, W, 1, 0, 1);
    conv(t0bf,   Wlf[k][1], ko[k].lb[1], nullptr, lfb, nullptr, 64, 49, H, W, 1, 0, 0); // lf
    chan_norm_kernel<<<dim3(gp(HW), 1, B), 256, 0, stream>>>(lfb, 49, 1, HW);
    mul_kernel<<<dim3((B * 64 * HW + 255) / 256), 256, 0, stream>>>(t1, t2, t1, B * 64 * HW);
    localfilter7_kernel<<<dim3(gp(HW), 64, B), 256, 0, stream>>>(t1, lfb, net, net_bf,
                                                                 64, H, W, 1.0f / W);
  }

  // ---- 7 residual blocks (residual add fused into 2nd conv) ----
  for (int k = 0; k < 7; ++k) {
    conv(net_bf, Wres[k][0], rb[k][0], nullptr, nullptr, t0bf, 64, 64, H, W, 3, 1, 1);
    conv(t0bf,   Wres[k][1], rb[k][1], net, net, net_bf, 64, 64, H, W, 3, 0, 0);
  }

  // ---- residual-image branch: conv -> relu -> PS2 -> conv -> PS2 -> conv ----
  conv(net_bf, Wsd[0], sb[0], nullptr, nullptr, big1bf, 64, 128, H, W, 3, 1, 1);
  pixel_shuffle2_bf_kernel<<<dim3(gp(192*192), 32, B), 256, 0, stream>>>(big1bf, big2bf, 32, H, W, 1.0f/192.f);
  conv(big2bf, Wsd[1], sb[1], nullptr, nullptr, big1bf, 32, 64, 192, 192, 3, 0, 0);
  pixel_shuffle2_bf_kernel<<<dim3(gp(384*384), 16, B), 256, 0, stream>>>(big1bf, big2bf, 16, 192, 192, 1.0f/384.f);
  conv(big2bf, Wsd[2], sb[2], nullptr, rbuf, nullptr, 16, 3, 384, 384, 3, 0, 0);

  // ---- kernel branch -> dynamic upsampling kernels ----
  conv(net_bf, Wkb[0], kbb[0], nullptr, nullptr, big1bf, 64, 128, H, W, 3, 1, 1);
  conv(big1bf, Wkb[1], kbb[1], nullptr, big0, nullptr, 128, 400, H, W, 3, 0, 0);
  chan_norm_kernel<<<dim3(gp(HW), 16, B), 256, 0, stream>>>(big0, 25, 16, HW);

  // ---- local 5x5 dynamic upsample (x4) + residual image -> d_out ----
  upsample_local_kernel<<<dim3(gp(HW), 3, B), 256, 0, stream>>>(lr, big0, rbuf,
                                                                (float*)d_out, H, W, 1.0f / W);
}
